// PairwiseScore_18700287607061
// MI455X (gfx1250) — matrix operand order repaired and checked by reading, not compiled, not run on hardware
//
#include <hip/hip_runtime.h>

// ---------------- problem constants ----------------
#define BB 2
#define NN 256
#define EE 512
#define HH 150
#define HPAD 160          // H padded to multiple of 16
#define HT 10             // number of 16-wide H tiles
#define KT1 16            // 512/32 k-steps, layer 1
#define KT2 5             // 160/32 k-steps, layer 2

// LDS strides (elements), padded for bank rotation, 16B aligned
#define A_STRIDE 520      // 512 + 8  bf16 -> 1040 B/row
#define H_STRIDE 168      // 160 + 8  bf16 -> 336 B/row
#define H2_STRIDE 164     // 160 + 4  f32  -> 656 B/row

typedef __bf16 v16bf __attribute__((ext_vector_type(16)));
typedef __bf16 v8bf  __attribute__((ext_vector_type(8)));
typedef float  v8f   __attribute__((ext_vector_type(8)));

__device__ __forceinline__ unsigned short f2bf(float f) {
  union { float f; unsigned u; } v; v.f = f;
  unsigned r = v.u + 0x7FFFu + ((v.u >> 16) & 1u);   // round-to-nearest-even
  return (unsigned short)(r >> 16);
}

// A-fragment: row pointer into a bf16 row, kbase = kt*32 + (lane>=16 ? 8 : 0)
__device__ __forceinline__ v16bf load_a_frag(const unsigned short* row, int kbase) {
  v8bf lo = *(const v8bf*)(const void*)(row + kbase);
  v8bf hi = *(const v8bf*)(const void*)(row + kbase + 16);
  return __builtin_shufflevector(lo, hi, 0,1,2,3,4,5,6,7,8,9,10,11,12,13,14,15);
}

// B-fragment from packed global layout: (((frag)*32 + lane)*16) contiguous bf16
__device__ __forceinline__ v16bf load_b_frag(const unsigned short* packed, int frag, int lane) {
  const v8bf* p = (const v8bf*)(const void*)(packed + (((size_t)frag * 32 + lane) << 4));
  v8bf lo = p[0];
  v8bf hi = p[1];
  return __builtin_shufflevector(lo, hi, 0,1,2,3,4,5,6,7,8,9,10,11,12,13,14,15);
}

// ---------------- pack W1c / W2 into WMMA-B fragment layout (bf16) ----------------
__global__ void pack_weights(const float* __restrict__ W1, const float* __restrict__ W2,
                             unsigned short* __restrict__ w1c_p, unsigned short* __restrict__ w2_p) {
  int idx = blockIdx.x * blockDim.x + threadIdx.x;   // 81920 threads
  {
    // layout: (((kt*HT + ht)*32 + lane)*16 + e), kt in [0,16)
    int e    = idx & 15;
    int lane = (idx >> 4) & 31;
    int ht   = (idx >> 9) % HT;
    int kt   = idx / (16 * 32 * HT);
    if (kt < KT1) {
      int k = kt * 32 + ((lane >= 16) ? 8 : 0) + ((e < 8) ? e : e + 8);   // < 512
      int n = ht * 16 + (lane & 15);                           // < 160
      float v = (n < HH) ? W1[(size_t)(2 * EE + k) * HH + n] : 0.f;
      w1c_p[idx] = f2bf(v);
    }
  }
  if (idx < KT2 * HT * 32 * 16) {
    int e    = idx & 15;
    int lane = (idx >> 4) & 31;
    int ht   = (idx >> 9) % HT;
    int kt   = idx / (16 * 32 * HT);
    int k = kt * 32 + ((lane >= 16) ? 8 : 0) + ((e < 8) ? e : e + 8);
    int n = ht * 16 + (lane & 15);
    float v = (k < HH && n < HH) ? W2[(size_t)k * HH + n] : 0.f;
    w2_p[idx] = f2bf(v);
  }
}

// ---------------- hi = g@W1a + b1 ; hj = g@W1b (tiny, VALU) ----------------
__global__ __launch_bounds__(HPAD) void hihj_kernel(const float* __restrict__ g,
                                                    const float* __restrict__ W1,
                                                    const float* __restrict__ b1,
                                                    float* __restrict__ hi_buf,
                                                    float* __restrict__ hj_buf) {
  __shared__ float gr[EE];
  int bn = blockIdx.x;                       // b*N + n
  for (int e = threadIdx.x; e < EE; e += blockDim.x)
    gr[e] = g[(size_t)bn * EE + e];
  __syncthreads();
  int h = threadIdx.x;
  float a1 = 0.f, a2 = 0.f;
  if (h < HH) {
    a1 = b1[h];
    for (int e = 0; e < EE; ++e) {
      float gv = gr[e];
      a1 += gv * W1[(size_t)e * HH + h];            // W1a
      a2 += gv * W1[(size_t)(EE + e) * HH + h];     // W1b
    }
  }
  hi_buf[(size_t)bn * HPAD + h] = a1;
  hj_buf[(size_t)bn * HPAD + h] = a2;
}

// ---------------- fused pairwise kernel: one block per (b,i), 10 waves ----------------
__global__ __launch_bounds__(320) void pairwise_main(
    const float* __restrict__ g, const float* __restrict__ m,
    const float* __restrict__ b2, const float* __restrict__ W3, const float* __restrict__ b3p,
    const unsigned short* __restrict__ w1c_p, const unsigned short* __restrict__ w2_p,
    const float* __restrict__ hi_buf, const float* __restrict__ hj_buf,
    float* __restrict__ out) {
  __shared__ __align__(16) unsigned short ldsA[16 * A_STRIDE];   // 16640 B
  __shared__ __align__(16) unsigned short ldsH[16 * H_STRIDE];   //  5376 B
  __shared__ __align__(16) float          ldsH2[16 * H2_STRIDE]; // 10496 B
  __shared__ float                        ldsGi[EE];             //  2048 B

  int bi = blockIdx.x;            // 0..511
  int b  = bi >> 8;
  int i  = bi & (NN - 1);

  for (int e = threadIdx.x; e < EE; e += blockDim.x)
    ldsGi[e] = g[(size_t)bi * EE + e];
  __syncthreads();

  const int lane    = threadIdx.x & 31;
  const int wave    = threadIdx.x >> 5;        // = h-tile index, 0..9
  const int col     = lane & 15;
  const int halfsel = lane >> 4;               // 0|1
  const int kbA     = halfsel * 8;             // A-fragment K sub-offset
  const int rowA    = col;                     // A-fragment row (M)
  const int hcol    = wave * 16 + col;         // this lane's output column

  const float hival = hi_buf[(size_t)bi * HPAD + hcol];   // includes b1
  const float b2v   = (hcol < HH) ? b2[hcol] : 0.f;
  const float b3v   = b3p[0];
  const float mi    = m[b * NN + i];

  for (int jt = 0; jt < NN / 16; ++jt) {
    const int j0 = jt * 16;

    // ---- build A tile: A[j][e] = g_i[e] * g_j[e] (f32 mul, bf16 store) ----
    for (int idx = threadIdx.x; idx < 16 * EE; idx += blockDim.x) {
      int j = idx >> 9;
      int e = idx & (EE - 1);
      float v = ldsGi[e] * g[(size_t)(b * NN + j0 + j) * EE + e];
      ldsA[j * A_STRIDE + e] = f2bf(v);
    }
    __syncthreads();

    // ---- layer 1: hij tile (16 j x 16 h), K = 512 ----
    v8f c1 = {};
    const unsigned short* arow = ldsA + rowA * A_STRIDE;
#pragma unroll
    for (int kt = 0; kt < KT1; ++kt) {
      v16bf a  = load_a_frag(arow, kt * 32 + kbA);
      v16bf bf = load_b_frag(w1c_p, kt * HT + wave, lane);
      c1 = __builtin_amdgcn_wmma_f32_16x16x32_bf16(false, a, false, bf, (short)0, c1, false, false);
    }
    // epilogue: + hi + hj, ReLU, bf16 -> LDS h-tile
#pragma unroll
    for (int v = 0; v < 8; ++v) {
      int row = v + 8 * halfsel;
      float hjv = hj_buf[(size_t)(b * NN + j0 + row) * HPAD + hcol];
      float x = c1[v] + hival + hjv;
      x = x > 0.f ? x : 0.f;
      ldsH[row * H_STRIDE + hcol] = f2bf(x);
    }
    __syncthreads();

    // ---- layer 2: h2 = relu(h @ W2 + b2), K = 160 ----
    v8f c2;
#pragma unroll
    for (int v = 0; v < 8; ++v) c2[v] = b2v;
    const unsigned short* hrow = ldsH + rowA * H_STRIDE;
#pragma unroll
    for (int kt = 0; kt < KT2; ++kt) {
      v16bf a  = load_a_frag(hrow, kt * 32 + kbA);
      v16bf bf = load_b_frag(w2_p, kt * HT + wave, lane);
      c2 = __builtin_amdgcn_wmma_f32_16x16x32_bf16(false, a, false, bf, (short)0, c2, false, false);
    }
#pragma unroll
    for (int v = 0; v < 8; ++v) {
      int row = v + 8 * halfsel;
      float x = c2[v];
      x = x > 0.f ? x : 0.f;
      ldsH2[row * H2_STRIDE + hcol] = x;
    }
    __syncthreads();

    // ---- layer 3: score = h2 @ W3 + b3 ; out = (m_i + m_j + score)/3 ----
    if (threadIdx.x < 256) {
      int j = threadIdx.x >> 4;
      int k = threadIdx.x & 15;
      float part = 0.f;
#pragma unroll
      for (int t = 0; t < HT; ++t) {
        int h = k + 16 * t;
        float w3 = (h < HH) ? W3[h] : 0.f;
        part += ldsH2[j * H2_STRIDE + h] * w3;
      }
      part += __shfl_xor(part, 8, 16);
      part += __shfl_xor(part, 4, 16);
      part += __shfl_xor(part, 2, 16);
      part += __shfl_xor(part, 1, 16);
      if (k == 0) {
        float score = part + b3v;
        float mj = m[b * NN + j0 + j];
        out[((size_t)(b * NN + i)) * NN + j0 + j] = (mi + mj + score) * (1.0f / 3.0f);
      }
    }
    __syncthreads();
  }
}

// ---------------- launcher ----------------
extern "C" void kernel_launch(void* const* d_in, const int* in_sizes, int n_in,
                              void* d_out, int out_size, void* d_ws, size_t ws_size,
                              hipStream_t stream) {
  const float* g  = (const float*)d_in[0];
  const float* m  = (const float*)d_in[1];
  const float* W1 = (const float*)d_in[2];
  const float* b1 = (const float*)d_in[3];
  const float* W2 = (const float*)d_in[4];
  const float* b2 = (const float*)d_in[5];
  const float* W3 = (const float*)d_in[6];
  const float* b3 = (const float*)d_in[7];
  float* out = (float*)d_out;

  char* ws = (char*)d_ws;
  unsigned short* w1c_p = (unsigned short*)ws;                              // 163840 B
  unsigned short* w2_p  = (unsigned short*)(ws + 163840);                   //  51200 B
  float* hi_buf = (float*)(ws + 163840 + 51200);                            // 327680 B
  float* hj_buf = (float*)(ws + 163840 + 51200 + 327680);                   // 327680 B

  pack_weights<<<320, 256, 0, stream>>>(W1, W2, w1c_p, w2_p);
  hihj_kernel<<<BB * NN, HPAD, 0, stream>>>(g, W1, b1, hi_buf, hj_buf);
  pairwise_main<<<BB * NN, 320, 0, stream>>>(g, m, b2, W3, b3,
                                             w1c_p, w2_p, hi_buf, hj_buf, out);
}
